// FusedMoEWithLoRA_36344013259314
// MI455X (gfx1250) — matrix operand order repaired and checked by reading, not compiled, not run on hardware
//
#include <hip/hip_runtime.h>
#include <math.h>

#define T_   1024
#define H_   2048
#define I_   1024
#define E_   16
#define KSEL 2
#define L_   4
#define R_   16
#define P_   (T_ * KSEL)
#define CAP_ P_
#define TM   32
#define KC   64
#define LDA  72   // bf16 elements per LDS tile row (64 data + 8 pad; keeps 16B alignment)

typedef __attribute__((ext_vector_type(16))) __bf16         v16bf;
typedef __attribute__((ext_vector_type(16))) unsigned short v16u;
typedef __attribute__((ext_vector_type(4)))  unsigned int   v4u;
typedef __attribute__((ext_vector_type(8)))  float          v8f;

// ---- fp32 -> bf16 round-half-up (1 add/elem), packed two-at-a-time with v_perm_b32 ----
static __device__ __forceinline__ unsigned int rhu(float f) {
  return __float_as_uint(f) + 0x8000u;
}
static __device__ __forceinline__ unsigned int pkbf(float a, float b) {
  // bytes 0-1 = hi16(rhu(a)), bytes 2-3 = hi16(rhu(b))
  return __builtin_amdgcn_perm(rhu(b), rhu(a), 0x07060302u);
}
// store float4 as 4 bf16 (8 bytes) into LDS
static __device__ __forceinline__ void st_bf4(unsigned short* dst, float4 v) {
  uint2 w;
  w.x = pkbf(v.x, v.y);
  w.y = pkbf(v.z, v.w);
  *(uint2*)dst = w;
}

// two 16-byte LDS loads -> one 16xbf16 fragment
struct frag_pair { v4u lo, hi; };
static __device__ __forceinline__ v16u ld2(const unsigned short* s, int o0, int o1) {
  frag_pair t{*(const v4u*)(s + o0), *(const v4u*)(s + o1)};
  return __builtin_bit_cast(v16u, t);
}
// A fragment: 16x32. lanes 0-15: row=lane, K {0..7,16..23}; lanes 16-31: row=lane-16, K {8..15,24..31}
static __device__ __forceinline__ v16u frag_a(const unsigned short* s, int row_base, int lane, int k0) {
  int base = (row_base + (lane & 15)) * LDA + k0 + ((lane >> 4) * 8);
  return ld2(s, base, base + 16);
}
// B fragment: 32x16. lanes 0-15: col=lane, K {0..7,16..23}; lanes 16-31: col=lane-16, K {8..15,24..31}
static __device__ __forceinline__ v16u frag_b(const unsigned short* s, int col_base, int lane, int k0) {
  int base = (col_base + (lane & 15)) * LDA + k0 + ((lane >> 4) * 8);
  return ld2(s, base, base + 16);
}

static __device__ __forceinline__ v8f wmma_bf16(v16u a, v16u b, v8f c) {
  return __builtin_amdgcn_wmma_f32_16x16x32_bf16(
      false, __builtin_bit_cast(v16bf, a),
      false, __builtin_bit_cast(v16bf, b),
      (short)0, c, false, false);
}

// ---------------- routing ----------------
__global__ __launch_bounds__(32) void zero_cnt_kernel(int* cnt) {
  if (threadIdx.x < E_) cnt[threadIdx.x] = 0;
}

__global__ __launch_bounds__(256) void route_kernel(const int* __restrict__ topk_ids,
                                                    int* __restrict__ cnt,
                                                    int* __restrict__ list) {
  int p = blockIdx.x * 256 + threadIdx.x;
  if (p < P_) {
    int e   = topk_ids[p];
    int pos = atomicAdd(&cnt[e], 1);
    list[e * CAP_ + pos] = p;
  }
}

// ---------------- LoRA: avec[p,r] = mask(r<rank)*scale * (x[t] . A[l,e,r,:]) ----------------
__global__ __launch_bounds__(256) void lora_a_kernel(const float* __restrict__ x,
                                                     const int* __restrict__ topk_ids,
                                                     const int* __restrict__ lora_idx,
                                                     const int* __restrict__ lora_rank,
                                                     const float* __restrict__ scalings,
                                                     const float* __restrict__ lora_a,
                                                     float* __restrict__ avec) {
  int p = blockIdx.x;
  int t = p >> 1;  // K = 2
  int e = topk_ids[p];
  int l = lora_idx[t];
  int tid = threadIdx.x, lane = tid & 31, wave = tid >> 5;
  const float4* x4 = (const float4*)(x + (size_t)t * H_);
  float sc = scalings[l];
  int   rk = lora_rank[l];
#pragma unroll
  for (int rr = 0; rr < 2; ++rr) {
    int r = wave + rr * 8;
    const float4* a4 = (const float4*)(lora_a + (((size_t)l * E_ + e) * R_ + r) * H_);
    float sum = 0.f;
    for (int j = lane; j < H_ / 4; j += 32) {
      float4 xv = x4[j], av = a4[j];
      sum += xv.x * av.x + xv.y * av.y + xv.z * av.z + xv.w * av.w;
    }
#pragma unroll
    for (int off = 16; off > 0; off >>= 1) sum += __shfl_down(sum, off, 32);
    if (lane == 0) avec[p * R_ + r] = (r < rk) ? sum * sc : 0.f;
  }
}

// ---------------- GEMM1: act[p, 0:I] = silu(x W1g^T) * (x W1u^T), grouped by expert ----------
__global__ __launch_bounds__(256) void gemm1_kernel(const float* __restrict__ x,
                                                    const float* __restrict__ w1,
                                                    const int* __restrict__ cnt,
                                                    const int* __restrict__ list,
                                                    float* __restrict__ act) {
  __shared__ unsigned short xs[TM * LDA];
  __shared__ unsigned short wg[64 * LDA];
  __shared__ unsigned short wu[64 * LDA];
  __shared__ int tok[TM];
  __shared__ int prow[TM];

  const int e     = blockIdx.y;
  const int count = cnt[e];
  const int m0    = blockIdx.x * TM;
  if (m0 >= count) return;
  const int n0   = blockIdx.z * 64;
  const int tid  = threadIdx.x;
  const int lane = tid & 31;
  const int wave = tid >> 5;
  const int wm   = wave & 1;
  const int wn   = wave >> 1;

  if (tid < TM) {
    int idx  = m0 + tid;
    int p    = (idx < count) ? list[e * CAP_ + idx] : -1;
    prow[tid] = p;
    tok[tid]  = (p >= 0) ? (p >> 1) : 0;
  }
  __syncthreads();

  v8f accG = {};
  v8f accU = {};

  const float* wgbase = w1 + ((size_t)e * 2 * I_ + n0) * H_;
  const float* wubase = w1 + ((size_t)e * 2 * I_ + I_ + n0) * H_;

  float4 xr[2], gr[4], ur[4];
  // prologue: load chunk 0 into registers
#pragma unroll
  for (int rep = 0; rep < 2; ++rep) {
    int v4 = tid + rep * 256, r = v4 >> 4, c4 = v4 & 15;
    xr[rep] = *(const float4*)(x + (size_t)tok[r] * H_ + c4 * 4);
  }
#pragma unroll
  for (int rep = 0; rep < 4; ++rep) {
    int v4 = tid + rep * 256, r = v4 >> 4, c4 = v4 & 15;
    gr[rep] = *(const float4*)(wgbase + (size_t)r * H_ + c4 * 4);
    ur[rep] = *(const float4*)(wubase + (size_t)r * H_ + c4 * 4);
  }

  for (int h0 = 0; h0 < H_; h0 += KC) {
    __syncthreads();  // previous chunk's fragment reads are done
#pragma unroll
    for (int rep = 0; rep < 2; ++rep) {
      int v4 = tid + rep * 256, r = v4 >> 4, c4 = v4 & 15;
      st_bf4(&xs[r * LDA + c4 * 4], xr[rep]);
    }
#pragma unroll
    for (int rep = 0; rep < 4; ++rep) {
      int v4 = tid + rep * 256, r = v4 >> 4, c4 = v4 & 15;
      st_bf4(&wg[r * LDA + c4 * 4], gr[rep]);
      st_bf4(&wu[r * LDA + c4 * 4], ur[rep]);
    }
    __syncthreads();
    const int h1 = h0 + KC;
    if (h1 < H_) {  // issue next chunk's global loads; consumed next iteration
#pragma unroll
      for (int rep = 0; rep < 2; ++rep) {
        int v4 = tid + rep * 256, r = v4 >> 4, c4 = v4 & 15;
        xr[rep] = *(const float4*)(x + (size_t)tok[r] * H_ + h1 + c4 * 4);
      }
#pragma unroll
      for (int rep = 0; rep < 4; ++rep) {
        int v4 = tid + rep * 256, r = v4 >> 4, c4 = v4 & 15;
        gr[rep] = *(const float4*)(wgbase + (size_t)r * H_ + h1 + c4 * 4);
        ur[rep] = *(const float4*)(wubase + (size_t)r * H_ + h1 + c4 * 4);
      }
    }
    // preload all fragments for the whole chunk, then issue WMMAs
    v16u a0  = frag_a(xs, wm * 16, lane, 0);
    v16u a1  = frag_a(xs, wm * 16, lane, 32);
    v16u bg0 = frag_b(wg, wn * 16, lane, 0);
    v16u bg1 = frag_b(wg, wn * 16, lane, 32);
    v16u bu0 = frag_b(wu, wn * 16, lane, 0);
    v16u bu1 = frag_b(wu, wn * 16, lane, 32);
    accG = wmma_bf16(a0, bg0, accG);
    accG = wmma_bf16(a1, bg1, accG);
    accU = wmma_bf16(a0, bu0, accU);
    accU = wmma_bf16(a1, bu1, accU);
  }

#pragma unroll
  for (int v = 0; v < 8; ++v) {
    int r = wm * 16 + v + 8 * (lane >> 4);
    int c = n0 + wn * 16 + (lane & 15);
    int p = prow[r];
    if (p >= 0) {
      float g = accG[v];
      float u = accU[v];
      float s = g / (1.0f + __expf(-g));  // silu
      act[(size_t)p * I_ + c] = s * u;
    }
  }
}

// ---------------- GEMM2: out_pair[p, 0:H] = topk_w[p] * (act[p] W2^T) ----------------
__global__ __launch_bounds__(256) void gemm2_kernel(const float* __restrict__ act,
                                                    const float* __restrict__ w2,
                                                    const float* __restrict__ topk_w,
                                                    const int* __restrict__ cnt,
                                                    const int* __restrict__ list,
                                                    float* __restrict__ out_pair) {
  __shared__ unsigned short as_[TM * LDA];
  __shared__ unsigned short wt[64 * LDA];
  __shared__ int prow[TM];

  const int e     = blockIdx.y;
  const int count = cnt[e];
  const int m0    = blockIdx.x * TM;
  if (m0 >= count) return;
  const int n0   = blockIdx.z * 64;  // over H_
  const int tid  = threadIdx.x;
  const int lane = tid & 31;
  const int wave = tid >> 5;
  const int wm   = wave & 1;
  const int wn   = wave >> 1;

  if (tid < TM) {
    int idx = m0 + tid;
    prow[tid] = (idx < count) ? list[e * CAP_ + idx] : -1;
  }
  __syncthreads();

  v8f acc = {};
  const float* wbase = w2 + ((size_t)e * H_ + n0) * I_;

  float4 ar[2], wr[4];
#pragma unroll
  for (int rep = 0; rep < 2; ++rep) {
    int v4 = tid + rep * 256, r = v4 >> 4, c4 = v4 & 15;
    int p = prow[r]; if (p < 0) p = 0;
    ar[rep] = *(const float4*)(act + (size_t)p * I_ + c4 * 4);
  }
#pragma unroll
  for (int rep = 0; rep < 4; ++rep) {
    int v4 = tid + rep * 256, r = v4 >> 4, c4 = v4 & 15;
    wr[rep] = *(const float4*)(wbase + (size_t)r * I_ + c4 * 4);
  }

  for (int k0 = 0; k0 < I_; k0 += KC) {
    __syncthreads();
#pragma unroll
    for (int rep = 0; rep < 2; ++rep) {
      int v4 = tid + rep * 256, r = v4 >> 4, c4 = v4 & 15;
      st_bf4(&as_[r * LDA + c4 * 4], ar[rep]);
    }
#pragma unroll
    for (int rep = 0; rep < 4; ++rep) {
      int v4 = tid + rep * 256, r = v4 >> 4, c4 = v4 & 15;
      st_bf4(&wt[r * LDA + c4 * 4], wr[rep]);
    }
    __syncthreads();
    const int k1 = k0 + KC;
    if (k1 < I_) {
#pragma unroll
      for (int rep = 0; rep < 2; ++rep) {
        int v4 = tid + rep * 256, r = v4 >> 4, c4 = v4 & 15;
        int p = prow[r]; if (p < 0) p = 0;
        ar[rep] = *(const float4*)(act + (size_t)p * I_ + k1 + c4 * 4);
      }
#pragma unroll
      for (int rep = 0; rep < 4; ++rep) {
        int v4 = tid + rep * 256, r = v4 >> 4, c4 = v4 & 15;
        wr[rep] = *(const float4*)(wbase + (size_t)r * I_ + k1 + c4 * 4);
      }
    }
    v16u a0 = frag_a(as_, wm * 16, lane, 0);
    v16u a1 = frag_a(as_, wm * 16, lane, 32);
    v16u b0 = frag_b(wt, wn * 16, lane, 0);
    v16u b1 = frag_b(wt, wn * 16, lane, 32);
    acc = wmma_bf16(a0, b0, acc);
    acc = wmma_bf16(a1, b1, acc);
  }

#pragma unroll
  for (int v = 0; v < 8; ++v) {
    int r = wm * 16 + v + 8 * (lane >> 4);
    int c = n0 + wn * 16 + (lane & 15);
    int p = prow[r];
    if (p >= 0) out_pair[(size_t)p * H_ + c] = topk_w[p] * acc[v];
  }
}

// ---------------- final: out[t] = sum_k out_pair + sum_k avec . B^T ----------------
__global__ __launch_bounds__(256) void final_kernel(const int* __restrict__ topk_ids,
                                                    const int* __restrict__ lora_idx,
                                                    const float* __restrict__ avec,
                                                    const float* __restrict__ lora_b,
                                                    const float* __restrict__ out_pair,
                                                    float* __restrict__ out) {
  __shared__ float av[KSEL][R_];
  __shared__ int ep[KSEL];
  int t   = blockIdx.x;
  int tid = threadIdx.x;
  if (tid < KSEL * R_) av[tid >> 4][tid & 15] = avec[(t * KSEL + (tid >> 4)) * R_ + (tid & 15)];
  if (tid < KSEL) ep[tid] = topk_ids[t * KSEL + tid];
  __syncthreads();
  int l = lora_idx[t];
  for (int h = tid; h < H_; h += 256) {
    float s = 0.f;
#pragma unroll
    for (int k = 0; k < KSEL; ++k) {
      int p = t * KSEL + k;
      s += out_pair[(size_t)p * H_ + h];
      const float4* b4 = (const float4*)(lora_b + (((size_t)l * E_ + ep[k]) * H_ + h) * R_);
      float4 b0 = b4[0], b1 = b4[1], b2 = b4[2], b3 = b4[3];
      s += av[k][0] * b0.x + av[k][1] * b0.y + av[k][2] * b0.z + av[k][3] * b0.w
         + av[k][4] * b1.x + av[k][5] * b1.y + av[k][6] * b1.z + av[k][7] * b1.w
         + av[k][8] * b2.x + av[k][9] * b2.y + av[k][10] * b2.z + av[k][11] * b2.w
         + av[k][12] * b3.x + av[k][13] * b3.y + av[k][14] * b3.z + av[k][15] * b3.w;
    }
    out[(size_t)t * H_ + h] = s;
  }
}

extern "C" void kernel_launch(void* const* d_in, const int* in_sizes, int n_in,
                              void* d_out, int out_size, void* d_ws, size_t ws_size,
                              hipStream_t stream) {
  const float* x         = (const float*)d_in[0];
  const int*   topk_ids  = (const int*)d_in[1];
  const float* topk_w    = (const float*)d_in[2];
  const int*   lora_idx  = (const int*)d_in[3];
  const int*   lora_rank = (const int*)d_in[4];
  const float* scalings  = (const float*)d_in[5];
  const float* w1        = (const float*)d_in[6];
  const float* w2        = (const float*)d_in[7];
  const float* lora_a    = (const float*)d_in[8];
  const float* lora_b    = (const float*)d_in[9];
  float* out = (float*)d_out;

  // workspace layout (bytes): cnt[16] @0, list[16*2048] @256, avec[2048*16] @131328,
  // act[2048*1024] @262400, out_pair[2048*2048] @8651008  -> total ~24.3 MB
  char*  ws       = (char*)d_ws;
  int*   cnt      = (int*)(ws);
  int*   list     = (int*)(ws + 256);
  float* avec     = (float*)(ws + 131328);
  float* act      = (float*)(ws + 262400);
  float* out_pair = (float*)(ws + 8651008);

  zero_cnt_kernel<<<1, 32, 0, stream>>>(cnt);
  route_kernel<<<P_ / 256, 256, 0, stream>>>(topk_ids, cnt, list);
  lora_a_kernel<<<P_, 256, 0, stream>>>(x, topk_ids, lora_idx, lora_rank, scalings, lora_a, avec);
  gemm1_kernel<<<dim3(CAP_ / TM, E_, I_ / 64), 256, 0, stream>>>(x, w1, cnt, list, act);
  gemm2_kernel<<<dim3(CAP_ / TM, E_, H_ / 64), 256, 0, stream>>>(act, w2, topk_w, cnt, list, out_pair);
  final_kernel<<<T_, 256, 0, stream>>>(topk_ids, lora_idx, avec, lora_b, out_pair, out);
}
